// NewMambaBlock_44461501448809
// MI455X (gfx1250) — compile-verified, matrix operand order checked by reading, not run on hardware
//
#include <hip/hip_runtime.h>
#include <hip/hip_bf16.h>
#include <math.h>

// ---- problem constants (match reference) ----
#define DMODEL 1024
#define EXP    2048          // E
#define NSTATE 16            // NS
#define DTRANK 64            // R
#define CONVK  4             // K
#define NB     4             // BB
#define SEQL   1024          // LL
#define MTOT   (NB*SEQL)     // 4096 rows
#define EPSV   1e-5f

typedef __bf16 bf16_t;
typedef __attribute__((ext_vector_type(16))) __bf16 v16bf;
typedef __attribute__((ext_vector_type(8)))  float  v8f;

// ---------------------------------------------------------------------------
// bf16 K-major fragment loader (A 16x32 and B 32x16 share this layout):
// lane l -> row l%16 ; chunks at K offsets (l/16)*8 and (l/16)*8+16
// ---------------------------------------------------------------------------
__device__ __forceinline__ v16bf load_frag_kmajor(const bf16_t* base, int ld) {
  const int lane = threadIdx.x & 31;
  const int row  = lane & 15;
  const int k0   = (lane >> 4) << 3;
  const bf16_t* p = base + (size_t)row * (size_t)ld + k0;
  union { v16bf v; uint4 q[2]; } r;
  r.q[0] = *reinterpret_cast<const uint4*>(p);
  r.q[1] = *reinterpret_cast<const uint4*>(p + 16);
  return r.v;
}

// fast silu: one v_rcp_f32 instead of the precise-division refinement chain
__device__ __forceinline__ float silu_f(float v) {
  return v * __builtin_amdgcn_rcpf(1.0f + __expf(-v));
}

// fast softplus: v_exp_f32 + v_log_f32
__device__ __forceinline__ float softplus_f(float t) {
  return (t > 20.0f) ? t : __logf(1.0f + __expf(t));
}

// CDNA5 async global->LDS copy (16B per lane), tracked by ASYNCcnt.
// LDS byte address = low 32 bits of the generic shared pointer (LDS aperture
// keeps the 0-based offset in addr[31:0]).
__device__ __forceinline__ void async_copy_b128(void* lds_dst, const void* gsrc) {
  unsigned l = (unsigned)(uintptr_t)lds_dst;
  asm volatile("global_load_async_to_lds_b128 %0, %1, off"
               :: "v"(l), "v"(gsrc) : "memory");
}
__device__ __forceinline__ void wait_asynccnt0() {
  asm volatile("s_wait_asynccnt 0x0" ::: "memory");
}

// ---------------------------------------------------------------------------
// Generic NT GEMM:  C[M,N] = A[M,K] * B[N,K]^T  (both K-contiguous, bf16 in,
// f32 accumulate via v_wmma_f32_16x16x32_bf16), fused per-mode epilogue.
// Wave tile: 64x32 (4x2 WMMA fragments -> 8 WMMA per 6 fragment loads).
// Double-buffered k-pipeline (two fragment sets, 64-wide k-steps) so fragment
// loads for step i+1 overlap the WMMAs of step i. KDIM: multiple of 64.
// MODE 0: inproj  -> n <  EXP : out1 (xc_raw f32, ld=EXP)
//                    n >= EXP : out2 (silu(z) f32, ld=EXP)
// MODE 1: xproj   -> out1 (xdbl f32, ld=96) + outb (xdbl bf16, ld=96)
// MODE 2: dt      -> out1 = softplus(acc + bias[n])  (f32, ld=EXP)
// MODE 3: outproj -> out1 (hidden f32, ld=DMODEL)
// ---------------------------------------------------------------------------
template <int MODE, int KDIM, int WGM, int WGN>
__global__ void __launch_bounds__(WGM*WGN*32)
gemm_nt_wmma(const bf16_t* __restrict__ A, int lda,
             const bf16_t* __restrict__ B, int ldb,
             float* __restrict__ out1, float* __restrict__ out2,
             const float* __restrict__ bias, bf16_t* __restrict__ outb)
{
  const int wave = threadIdx.x >> 5;
  const int lane = threadIdx.x & 31;
  const int wm = wave % WGM;
  const int wn = wave / WGM;
  const int m0 = blockIdx.y * (WGM*64) + wm*64;
  const int n0 = blockIdx.x * (WGN*32) + wn*32;

  v8f acc[4][2] = {};

  v16bf af0[4], bf0[2], af1[4], bf1[2];

  auto load_set = [&](int k0, v16bf af[4], v16bf bf[2]) {
    #pragma unroll
    for (int i = 0; i < 4; ++i)
      af[i] = load_frag_kmajor(A + (size_t)(m0 + 16*i) * lda + k0, lda);
    #pragma unroll
    for (int j = 0; j < 2; ++j)
      bf[j] = load_frag_kmajor(B + (size_t)(n0 + 16*j) * ldb + k0, ldb);
  };
  auto mmas = [&](v16bf af[4], v16bf bf[2]) {
    #pragma unroll
    for (int i = 0; i < 4; ++i)
      #pragma unroll
      for (int j = 0; j < 2; ++j)
        acc[i][j] = __builtin_amdgcn_wmma_f32_16x16x32_bf16(
            false, af[i], false, bf[j], (short)0, acc[i][j], false, false);
  };

  load_set(0, af0, bf0);
  #pragma unroll 1
  for (int k0 = 0; k0 < KDIM - 64; k0 += 64) {
    load_set(k0 + 32, af1, bf1);
    mmas(af0, bf0);
    load_set(k0 + 64, af0, bf0);
    mmas(af1, bf1);
  }
  // tail step (k0 = KDIM-64)
  load_set(KDIM - 32, af1, bf1);
  mmas(af0, bf0);
  mmas(af1, bf1);

  // C/D layout: lane n = lane&15 ; VGPR r holds M = r + 8*(lane>>4)
  const int cn = lane & 15;
  const int cm = (lane >> 4) * 8;

  #pragma unroll
  for (int i = 0; i < 4; ++i) {
    #pragma unroll
    for (int j = 0; j < 2; ++j) {
      const int gm0 = m0 + i*16 + cm;
      const int gn  = n0 + j*16 + cn;
      #pragma unroll
      for (int r = 0; r < 8; ++r) {
        const int gm = gm0 + r;
        const float v = acc[i][j][r];
        if (MODE == 0) {
          if (gn < EXP) out1[(size_t)gm*EXP + gn] = v;
          else          out2[(size_t)gm*EXP + (gn - EXP)] = silu_f(v);
        } else if (MODE == 1) {
          out1[(size_t)gm*96 + gn] = v;
          outb[(size_t)gm*96 + gn] = (bf16_t)v;
        } else if (MODE == 2) {
          out1[(size_t)gm*EXP + gn] = softplus_f(v + bias[gn]);
        } else {
          out1[(size_t)gm*DMODEL + gn] = v;
        }
      }
    }
  }
}

// ---------------------------------------------------------------------------
// f32 -> bf16 weight conversion
// ---------------------------------------------------------------------------
__global__ void __launch_bounds__(256)
f32_to_bf16_kernel(const float* __restrict__ in, bf16_t* __restrict__ out, int n) {
  const int i = blockIdx.x * 256 + threadIdx.x;
  if (i < n) out[i] = (bf16_t)in[i];
}

// ---------------------------------------------------------------------------
// rmsnorm(x[b, lsrc]) * w -> bf16 row m   (flip selects time reversal)
// ---------------------------------------------------------------------------
__global__ void __launch_bounds__(256)
rmsnorm_in_kernel(const float* __restrict__ x, const float* __restrict__ w,
                  bf16_t* __restrict__ xn, int flip)
{
  const int m = blockIdx.x;
  const int b = m / SEQL;
  const int l = m % SEQL;
  const int lsrc = flip ? (SEQL - 1 - l) : l;
  const float* row = x + ((size_t)b * SEQL + lsrc) * DMODEL;

  float v[4];
  float ss = 0.0f;
  #pragma unroll
  for (int i = 0; i < 4; ++i) {
    v[i] = row[threadIdx.x + i*256];
    ss += v[i] * v[i];
  }
  __shared__ float red[256];
  red[threadIdx.x] = ss;
  __syncthreads();
  for (int s = 128; s > 0; s >>= 1) {
    if (threadIdx.x < s) red[threadIdx.x] += red[threadIdx.x + s];
    __syncthreads();
  }
  const float scale = rsqrtf(red[0] * (1.0f/(float)DMODEL) + EPSV);
  bf16_t* o = xn + (size_t)m * DMODEL;
  #pragma unroll
  for (int i = 0; i < 4; ++i) {
    const int c = threadIdx.x + i*256;
    o[c] = (bf16_t)(v[i] * scale * w[c]);
  }
}

// ---------------------------------------------------------------------------
// causal depthwise conv (K=4) + bias + silu : xc_raw(f32) -> xc(bf16)
// ---------------------------------------------------------------------------
__global__ void __launch_bounds__(256)
conv_silu_kernel(const float* __restrict__ xcraw, const float* __restrict__ cw,
                 const float* __restrict__ cb, bf16_t* __restrict__ xcb)
{
  const size_t i = (size_t)blockIdx.x * 256 + threadIdx.x;   // < MTOT*EXP
  const int e = (int)(i % EXP);
  const size_t m = i / EXP;
  const int l = (int)(m % SEQL);
  float acc = cb[e];
  #pragma unroll
  for (int k = 0; k < CONVK; ++k) {
    const int lk = l - (CONVK - 1) + k;
    if (lk >= 0)
      acc += cw[e*CONVK + k] * xcraw[(m - (size_t)(CONVK - 1 - k)) * EXP + e];
  }
  xcb[i] = (bf16_t)silu_f(acc);
}

// ---------------------------------------------------------------------------
// Sequential SSM scan: one lane per (b, e), 16 states in registers.
// B/C (shared across e) staged to LDS in 64-step chunks with CDNA5
// global_load_async_to_lds_b128 (ASYNCcnt) — xdbl cols 64..95 are 128B
// contiguous per timestep, matching sBC rows.
// Fused: y = (scan + xc*Dp) * silu(z)  -> bf16 for outproj.
// ---------------------------------------------------------------------------
#define TCH 64
__global__ void __launch_bounds__(256)
ssm_scan_kernel(const float* __restrict__ dtf, const bf16_t* __restrict__ xcb,
                const float* __restrict__ szf, const float* __restrict__ xdbl,
                const float* __restrict__ Alog, const float* __restrict__ Dp,
                bf16_t* __restrict__ yg)
{
  const int b  = blockIdx.x >> 3;           // EXP/256 = 8 groups per batch
  const int eg = blockIdx.x & 7;
  const int e  = eg * 256 + threadIdx.x;

  float A[NSTATE];
  #pragma unroll
  for (int n = 0; n < NSTATE; ++n) A[n] = -__expf(Alog[(size_t)e*NSTATE + n]);
  const float dcoef = Dp[e];

  float h[NSTATE];
  #pragma unroll
  for (int n = 0; n < NSTATE; ++n) h[n] = 0.0f;

  __shared__ float sBC[TCH][2*NSTATE];   // [t][0..15]=B, [t][16..31]=C ; 8KB

  for (int t0 = 0; t0 < SEQL; t0 += TCH) {
    // stage B/C for TCH timesteps: 512 x 16B async transfers, 2 per thread
    #pragma unroll
    for (int it = 0; it < 2; ++it) {
      const int i  = threadIdx.x + it*256;     // 0..511
      const int tt = i >> 3;                   // timestep in chunk
      const int q  = i & 7;                    // 16B chunk within 128B row
      const float* g = xdbl + ((size_t)(b*SEQL + t0 + tt)) * 96 + DTRANK + q*4;
      async_copy_b128(&sBC[tt][q*4], g);
    }
    wait_asynccnt0();
    __syncthreads();

    for (int tt = 0; tt < TCH; ++tt) {
      const size_t idx = ((size_t)(b*SEQL + t0 + tt)) * EXP + e;
      const float dtv = dtf[idx];
      const float xv  = (float)xcb[idx];
      const float du  = dtv * xv;
      float y = 0.0f;
      #pragma unroll
      for (int n = 0; n < NSTATE; ++n) {
        const float dA = __expf(dtv * A[n]);
        h[n] = dA * h[n] + du * sBC[tt][n];
        y += h[n] * sBC[tt][NSTATE + n];
      }
      y = (y + xv * dcoef) * szf[idx];
      yg[idx] = (bf16_t)y;
    }
    __syncthreads();
  }
}

// ---------------------------------------------------------------------------
// final: rmsnorm(hidden + residual, normf), store (fwd) or add-flipped (bwd)
// ---------------------------------------------------------------------------
__global__ void __launch_bounds__(256)
final_kernel(const float* __restrict__ x, const float* __restrict__ hidden,
             const float* __restrict__ normf, float* __restrict__ out,
             int flip, int accumulate)
{
  const int m = blockIdx.x;
  const int b = m / SEQL;
  const int l = m % SEQL;
  const int lsrc = flip ? (SEQL - 1 - l) : l;
  const float* xr = x + ((size_t)b * SEQL + lsrc) * DMODEL;
  const float* hr = hidden + (size_t)m * DMODEL;

  float v[4];
  float ss = 0.0f;
  #pragma unroll
  for (int i = 0; i < 4; ++i) {
    const int c = threadIdx.x + i*256;
    v[i] = hr[c] + xr[c];
    ss += v[i] * v[i];
  }
  __shared__ float red[256];
  red[threadIdx.x] = ss;
  __syncthreads();
  for (int s = 128; s > 0; s >>= 1) {
    if (threadIdx.x < s) red[threadIdx.x] += red[threadIdx.x + s];
    __syncthreads();
  }
  const float scale = rsqrtf(red[0] * (1.0f/(float)DMODEL) + EPSV);
  float* orow = out + ((size_t)b * SEQL + lsrc) * DMODEL;
  #pragma unroll
  for (int i = 0; i < 4; ++i) {
    const int c = threadIdx.x + i*256;
    const float r = v[i] * scale * normf[c];
    if (accumulate) orow[c] += r;
    else            orow[c]  = r;
  }
}

// ---------------------------------------------------------------------------
// host launch
// ---------------------------------------------------------------------------
extern "C" void kernel_launch(void* const* d_in, const int* in_sizes, int n_in,
                              void* d_out, int out_size, void* d_ws, size_t ws_size,
                              hipStream_t stream)
{
  (void)in_sizes; (void)n_in; (void)out_size; (void)ws_size;

  const float* x     = (const float*)d_in[0];
  const float* normf = (const float*)d_in[21];
  char* ws = (char*)d_ws;

  constexpr size_t MB = 1ull << 20;
  // workspace layout (reused across the two direction passes), ~153 MB:
  bf16_t* xn    = (bf16_t*)(ws + 0*MB);      //  8 MB  bf16 rmsnormed input
  float*  xcraw = (float*)(ws + 8*MB);       // 32 MB  pre-conv xc (f32)
  float*  hidden= xcraw;                     // alias: xcraw is dead after conv
  float*  szf   = (float*)(ws + 40*MB);      // 32 MB  silu(z) (f32)
  bf16_t* xcb   = (bf16_t*)(ws + 72*MB);     // 16 MB  post-conv xc (bf16)
  float*  xdbl  = (float*)(ws + 88*MB);      //  2 MB  xdbl f32 (ld 96)
  bf16_t* xdblb = (bf16_t*)(ws + 90*MB);     //  1 MB  xdbl bf16 (ld 96)
  float*  dtf   = (float*)(ws + 91*MB);      // 32 MB  softplus dt (f32)
  bf16_t* yg    = (bf16_t*)(ws + 123*MB);    // 16 MB  gated y (bf16)
  bf16_t* inpb  = (bf16_t*)(ws + 139*MB);    //  8 MB  inproj bf16
  bf16_t* xprb  = (bf16_t*)(ws + 147*MB);    //  1 MB  xproj bf16
  bf16_t* dtwb  = (bf16_t*)(ws + 148*MB);    //  1 MB  dtw bf16
  bf16_t* outpb = (bf16_t*)(ws + 149*MB);    //  4 MB  outproj bf16

  for (int dir = 0; dir < 2; ++dir) {
    const float* nw      = (const float*)d_in[ 1 + dir*10];
    const float* inproj  = (const float*)d_in[ 2 + dir*10];
    const float* convw   = (const float*)d_in[ 3 + dir*10];
    const float* convb   = (const float*)d_in[ 4 + dir*10];
    const float* xproj   = (const float*)d_in[ 5 + dir*10];
    const float* dtw     = (const float*)d_in[ 6 + dir*10];
    const float* dtb     = (const float*)d_in[ 7 + dir*10];
    const float* Alog    = (const float*)d_in[ 8 + dir*10];
    const float* Dp      = (const float*)d_in[ 9 + dir*10];
    const float* outproj = (const float*)d_in[10 + dir*10];

    // weight conversion to bf16
    {
      int n;
      n = 2*EXP*DMODEL;   f32_to_bf16_kernel<<<(n+255)/256,256,0,stream>>>(inproj,  inpb,  n);
      n = 96*EXP;         f32_to_bf16_kernel<<<(n+255)/256,256,0,stream>>>(xproj,   xprb,  n);
      n = EXP*DTRANK;     f32_to_bf16_kernel<<<(n+255)/256,256,0,stream>>>(dtw,     dtwb,  n);
      n = DMODEL*EXP;     f32_to_bf16_kernel<<<(n+255)/256,256,0,stream>>>(outproj, outpb, n);
    }

    // 1) rmsnorm (+flip for bwd) -> bf16
    rmsnorm_in_kernel<<<MTOT, 256, 0, stream>>>(x, nw, xn, dir);

    // 2) inproj GEMM: [4096,4096] = xn[4096,1024] * inproj^T ; split xc/silu(z)
    gemm_nt_wmma<0,DMODEL,2,4><<<dim3((2*EXP)/128, MTOT/128), 256, 0, stream>>>(
        xn, DMODEL, inpb, DMODEL, xcraw, szf, nullptr, nullptr);

    // 3) causal conv + silu -> xc bf16
    conv_silu_kernel<<<(MTOT*EXP)/256, 256, 0, stream>>>(xcraw, convw, convb, xcb);

    // 4) xproj GEMM: [4096,96] = xc[4096,2048] * xproj^T
    gemm_nt_wmma<1,EXP,8,1><<<dim3(96/32, MTOT/512), 256, 0, stream>>>(
        xcb, EXP, xprb, EXP, xdbl, nullptr, nullptr, xdblb);

    // 5) dt GEMM: [4096,2048] = xdbl[:, :64] * dtw^T ; softplus(+dtb)
    gemm_nt_wmma<2,DTRANK,2,4><<<dim3(EXP/128, MTOT/128), 256, 0, stream>>>(
        xdblb, 96, dtwb, DTRANK, dtf, nullptr, dtb, nullptr);

    // 6) sequential SSM scan + D-term + silu(z) gating -> bf16
    ssm_scan_kernel<<<NB*(EXP/256), 256, 0, stream>>>(dtf, xcb, szf, xdbl, Alog, Dp, yg);

    // 7) outproj GEMM: [4096,1024] = yg[4096,2048] * outproj^T -> hidden
    gemm_nt_wmma<3,EXP,2,4><<<dim3(DMODEL/128, MTOT/128), 256, 0, stream>>>(
        yg, EXP, outpb, EXP, hidden, nullptr, nullptr, nullptr);

    // 8) residual + rmsnorm(normf); fwd stores, bwd accumulates at flipped l
    final_kernel<<<MTOT, 256, 0, stream>>>(x, hidden, normf, (float*)d_out, dir, dir);
  }
}
